// SlotMemory_65618510348344
// MI455X (gfx1250) — matrix operand order repaired and checked by reading, not compiled, not run on hardware
//
#include <hip/hip_runtime.h>
#include <hip/hip_bf16.h>
#include <math.h>

typedef __attribute__((ext_vector_type(16))) __bf16 v16bf;
typedef __attribute__((ext_vector_type(8)))  __bf16 v8bf;
typedef __attribute__((ext_vector_type(8)))  float  v8f;
typedef __attribute__((ext_vector_type(4)))  float  v4f;

#define B_     4
#define S_     4096
#define H_     1024
#define M_     128
#define NS_    32
#define CS_    64
#define NC_    64
#define ROWS_  (B_ * S_)          // 16384
#define DECAY_C 0.93797493f       // 0.999^64
#define INV_SCALE 0.08838834764831845f  // 1/sqrt(128)

// ---------------------------------------------------------------------------
// Fragment layouts in global memory (bf16, 32B per lane, k contiguous per lane half):
//  A-frag:  a[rt][kt][lane][e]  lane = half*16 + (row&15)
//           K = kt*32 + half*16 + e
//  B-frag:  b[kt][nt][lane][e]  lane = kg*16 + n ; K = kt*32 + kg*16 + e
// ---------------------------------------------------------------------------

// Convert row-major f32 (rows x Kdim) into A-fragment bf16 layout.
// One thread handles one K-octet of one row (8 contiguous elements).
__global__ __launch_bounds__(256) void k_prep_a(const float* __restrict__ src,
                                                __bf16* __restrict__ dst,
                                                int Kdim, int oshift, int ktiles) {
  const int oid = blockIdx.x * 256 + threadIdx.x;
  const int row = oid >> oshift;
  const int oc  = oid & ((1 << oshift) - 1);
  const int k0  = oc * 8;
  const int kt  = k0 >> 5;
  const int c   = k0 & 31;
  const int half = c >> 4;
  const int e0   = c & 8;
  const int rt = row >> 4, m16 = row & 15;
  const int lane = half * 16 + m16;

  v4f f0 = *(const v4f*)(src + (size_t)row * Kdim + k0);
  v4f f1 = *(const v4f*)(src + (size_t)row * Kdim + k0 + 4);
  v8bf v;
#pragma unroll
  for (int j = 0; j < 4; ++j) { v[j] = (__bf16)f0[j]; v[j + 4] = (__bf16)f1[j]; }
  size_t di = ((size_t)(rt * ktiles + kt) * 32 + lane) * 16 + e0;
  *(v8bf*)(dst + di) = v;
}

// Convert row-major f32 (K x N) weight into B-fragment bf16 layout.
// One thread handles 8 consecutive e (8 K-rows) for one (kt,nt,lane).
__global__ __launch_bounds__(256) void k_prep_b(const float* __restrict__ src,
                                                __bf16* __restrict__ dst,
                                                int N, int ntshift) {
  const int o = blockIdx.x * 256 + threadIdx.x;   // octet index
  const int e0   = (o & 1) * 8;
  const int lane = (o >> 1) & 31;
  const int rest = o >> 6;
  const int nt = rest & ((1 << ntshift) - 1);
  const int kt = rest >> ntshift;
  const int kg = lane >> 4, n = lane & 15;
  const int col = nt * 16 + n;
  const int rbase = kt * 32 + kg * 16 + e0;
  v8bf v;
#pragma unroll
  for (int j = 0; j < 8; ++j) v[j] = (__bf16)src[(size_t)(rbase + j) * N + col];
  *(v8bf*)(dst + (size_t)o * 8) = v;
}

// ---------------------------------------------------------------------------
// Kernel 1: projections q,k,v,wk = x @ W via WMMA on pre-swizzled bf16 frags.
// grid = (ROWS_/128, 4), block = 256 (wave w owns row-tile rt = bx*8+w).
// No LDS, no barriers: 32B vector loads feed v_wmma directly.
// ---------------------------------------------------------------------------
__global__ __launch_bounds__(256) void k_proj(const __bf16* __restrict__ xbf,
                                              const __bf16* __restrict__ wbf_all,
                                              float* __restrict__ ws) {
  const int proj = blockIdx.y;
  const __bf16* wbf = wbf_all + (size_t)proj * (H_ * M_);
  float* Out = ws + (size_t)proj * (ROWS_ * M_);

  const int tid  = threadIdx.x;
  const int wave = tid >> 5, lane = tid & 31;
  const int half = lane >> 4, m16 = lane & 15;
  const int rt = blockIdx.x * 8 + wave;

  v8f zero = {};
  v8f acc[8];
#pragma unroll
  for (int i = 0; i < 8; ++i) acc[i] = zero;

  for (int kt = 0; kt < 32; ++kt) {
    v16bf a = *(const v16bf*)(xbf + (((size_t)(rt * 32 + kt) * 32 + lane) << 4));
    if (kt + 1 < 32)
      __builtin_prefetch(xbf + (((size_t)(rt * 32 + kt + 1) * 32 + lane) << 4), 0, 0);
#pragma unroll
    for (int nt = 0; nt < 8; ++nt) {
      v16bf b = *(const v16bf*)(wbf + (((size_t)(kt * 8 + nt) * 32 + lane) << 4));
      acc[nt] = __builtin_amdgcn_wmma_f32_16x16x32_bf16(
          false, a, false, b, (short)0, acc[nt], false, false);
    }
  }

  if (proj != 0) {  // fused row L2-normalize for k / v / wk
    float inv[8];
#pragma unroll
    for (int r = 0; r < 8; ++r) {
      float s = 0.f;
#pragma unroll
      for (int nt = 0; nt < 8; ++nt) s += acc[nt][r] * acc[nt][r];
      s += __shfl_xor(s, 1, 32);
      s += __shfl_xor(s, 2, 32);
      s += __shfl_xor(s, 4, 32);
      s += __shfl_xor(s, 8, 32);
      inv[r] = 1.0f / sqrtf(fmaxf(s, 1e-24f));
    }
#pragma unroll
    for (int nt = 0; nt < 8; ++nt)
#pragma unroll
      for (int r = 0; r < 8; ++r) acc[nt][r] *= inv[r];
  }

#pragma unroll
  for (int nt = 0; nt < 8; ++nt)
#pragma unroll
    for (int r = 0; r < 8; ++r) {
      int row = rt * 16 + r + 8 * half;
      Out[(size_t)row * M_ + nt * 16 + m16] = acc[nt][r];
    }
}

// ---------------------------------------------------------------------------
// Kernel 2: gate g = sigmoid(x @ Wg_w + Wg_b). One wave per row.
// ---------------------------------------------------------------------------
__global__ __launch_bounds__(256) void k_gate(const float* __restrict__ X,
                                              const float* __restrict__ Wg,
                                              const float* __restrict__ Wb,
                                              float* __restrict__ g) {
  const int wave = threadIdx.x >> 5, lane = threadIdx.x & 31;
  const int row = blockIdx.x * 8 + wave;
  float s = 0.f;
  for (int j = lane; j < H_; j += 32) s += X[(size_t)row * H_ + j] * Wg[j];
  s += __shfl_xor(s, 1, 32);
  s += __shfl_xor(s, 2, 32);
  s += __shfl_xor(s, 4, 32);
  s += __shfl_xor(s, 8, 32);
  s += __shfl_xor(s, 16, 32);
  if (lane == 0) g[row] = 1.0f / (1.0f + expf(-(s + Wb[0])));
}

// ---------------------------------------------------------------------------
// Kernel 3: sequential slot-memory scan (64 chunk steps) in FP32.
// Single workgroup of 256 threads carries slot state in LDS.
// ---------------------------------------------------------------------------
__global__ __launch_bounds__(256) void k_scan(const float* __restrict__ Q,
                                              const float* __restrict__ K,
                                              const float* __restrict__ V,
                                              const float* __restrict__ WK,
                                              const float* __restrict__ G,
                                              const float* __restrict__ SK0,
                                              const float* __restrict__ SV0,
                                              const float* __restrict__ SU0,
                                              float* __restrict__ R,
                                              float* __restrict__ out_scalar) {
  __shared__ float sk_s[NS_][M_];
  __shared__ float sv_s[NS_][M_];
  __shared__ float su_s[NS_];
  __shared__ float g_s[256];
  __shared__ float kagg[M_], vagg[M_], wkagg[M_];
  __shared__ float esc[NS_];
  __shared__ float sh_ws_mean, sh_su0, sh_sud;

  const int tid = threadIdx.x;
  for (int i = tid; i < NS_ * M_; i += 256) {
    sk_s[i >> 7][i & 127] = SK0[i];
    sv_s[i >> 7][i & 127] = SV0[i];
  }
  if (tid < NS_) su_s[tid] = SU0[tid];
  __syncthreads();

  float wss_sum = 0.f;
  const int bb = tid >> 6, cc = tid & 63;

  for (int t = 0; t < NC_; ++t) {
    const int grow = bb * S_ + t * CS_ + cc;
    g_s[tid] = G[grow];
    __syncthreads();

    if (tid == 0) {
      float s0 = 0.f;
      for (int n = 0; n < NS_; ++n) s0 += su_s[n];
      sh_su0 = s0;
      float gs = 0.f;
      for (int i = 0; i < 256; ++i) gs += g_s[i];
      sh_ws_mean = gs * (1.0f / 256.0f);
    }
    __syncthreads();

    // scores = q . sk^T / sqrt(M) + mask ; softmax
    float sc[NS_];
#pragma unroll
    for (int n = 0; n < NS_; ++n) sc[n] = 0.f;
    for (int kb = 0; kb < 4; ++kb) {
      float qv[32];
#pragma unroll
      for (int j = 0; j < 32; ++j) qv[j] = Q[(size_t)grow * M_ + kb * 32 + j];
#pragma unroll
      for (int n = 0; n < NS_; ++n) {
        float a = 0.f;
#pragma unroll
        for (int j = 0; j < 32; ++j) a += qv[j] * sk_s[n][kb * 32 + j];
        sc[n] += a;
      }
    }
    float mx = -1e30f;
#pragma unroll
    for (int n = 0; n < NS_; ++n) {
      sc[n] = sc[n] * INV_SCALE + ((su_s[n] > 1e-8f) ? 0.0f : -1e9f);
      mx = fmaxf(mx, sc[n]);
    }
    float den = 0.f;
#pragma unroll
    for (int n = 0; n < NS_; ++n) {
      sc[n] = expf(sc[n] - mx);
      den += sc[n];
    }
    float rden = 1.0f / den;
#pragma unroll
    for (int n = 0; n < NS_; ++n) sc[n] *= rden;

    // r = attn @ sv (zeroed if memory empty)
    const bool empty = (sh_su0 < 1e-8f);
    for (int mb = 0; mb < 4; ++mb) {
      float rv[32];
#pragma unroll
      for (int j = 0; j < 32; ++j) rv[j] = 0.f;
#pragma unroll
      for (int n = 0; n < NS_; ++n) {
        float a = sc[n];
#pragma unroll
        for (int j = 0; j < 32; ++j) rv[j] += a * sv_s[n][mb * 32 + j];
      }
#pragma unroll
      for (int j = 0; j < 32; ++j)
        R[(size_t)grow * M_ + mb * 32 + j] = empty ? 0.0f : rv[j];
    }
    __syncthreads();

    // gated aggregates over (batch,chunk); w_sum absorbed by normalize
    if (tid < M_) {
      const int m = tid;
      float ka = 0.f, va = 0.f, wa = 0.f;
      for (int row = 0; row < 256; ++row) {
        float w = g_s[row];
        int gr = (row >> 6) * S_ + t * CS_ + (row & 63);
        ka += K[(size_t)gr * M_ + m] * w;
        va += V[(size_t)gr * M_ + m] * w;
        wa += WK[(size_t)gr * M_ + m] * w;
      }
      kagg[m] = ka; vagg[m] = va; wkagg[m] = wa;
    }
    __syncthreads();
    if (tid < 3) {
      float* a = (tid == 0) ? kagg : (tid == 1) ? vagg : wkagg;
      float s2 = 0.f;
      for (int m = 0; m < M_; ++m) s2 += a[m] * a[m];
      float inv = 1.0f / sqrtf(fmaxf(s2, 1e-24f));
      for (int m = 0; m < M_; ++m) a[m] *= inv;
    }
    __syncthreads();

    // write: decay usage, slot scoring
    if (tid < NS_) {
      const int n = tid;
      float sun = su_s[n] * DECAY_C;
      su_s[n] = sun;
      float sim = 0.f;
      for (int m = 0; m < M_; ++m) sim += sk_s[n][m] * wkagg[m];
      float emptiness = 1.0f - fminf(fmaxf(sun, 0.0f), 1.0f);
      esc[n] = (sim + emptiness * 2.0f) * 5.0f;
    }
    __syncthreads();
    if (tid == 0) {
      float s = 0.f;
      for (int n = 0; n < NS_; ++n) s += su_s[n];
      sh_sud = s;
    }
    __syncthreads();
    if (tid < NS_) {
      const int n = tid;
      float emx = -1e30f;
      for (int j = 0; j < NS_; ++j) emx = fmaxf(emx, esc[j]);
      float eden = 0.f;
      for (int j = 0; j < NS_; ++j) eden += expf(esc[j] - emx);
      float wwn = expf(esc[n] - emx) / eden;
      if (sh_sud < 1e-8f) wwn = 1.0f / (float)NS_;
      const float wsm = sh_ws_mean;
      const float alpha = wsm * wwn;

      float s2 = 0.f;
      for (int m = 0; m < M_; ++m) {
        float tv = alpha * kagg[m] + (1.0f - alpha) * sk_s[n][m];
        sk_s[n][m] = tv; s2 += tv * tv;
      }
      float inv = 1.0f / sqrtf(fmaxf(s2, 1e-24f));
      for (int m = 0; m < M_; ++m) sk_s[n][m] *= inv;

      s2 = 0.f;
      for (int m = 0; m < M_; ++m) {
        float tv = alpha * vagg[m] + (1.0f - alpha) * sv_s[n][m];
        sv_s[n][m] = tv; s2 += tv * tv;
      }
      inv = 1.0f / sqrtf(fmaxf(s2, 1e-24f));
      for (int m = 0; m < M_; ++m) sv_s[n][m] *= inv;

      su_s[n] += wwn * wsm;
    }
    if (tid == 0) wss_sum += sh_ws_mean;
    __syncthreads();
  }

  if (tid == 0) out_scalar[0] = wss_sum * (1.0f / (float)NC_);
}

// ---------------------------------------------------------------------------
// Kernel 4: out = retrieved @ Wout via WMMA on pre-swizzled frags,
// fused per-row sumsq for the norm-10 clamp. grid = ROWS_/128, block = 256.
// ---------------------------------------------------------------------------
__global__ __launch_bounds__(256) void k_out(const __bf16* __restrict__ rbf,
                                             const __bf16* __restrict__ wobf,
                                             float* __restrict__ Out,
                                             float* __restrict__ rowscale) {
  const int tid  = threadIdx.x;
  const int wave = tid >> 5, lane = tid & 31;
  const int half = lane >> 4, m16 = lane & 15;
  const int rt = blockIdx.x * 8 + wave;

  v16bf af[4];
#pragma unroll
  for (int kb = 0; kb < 4; ++kb)
    af[kb] = *(const v16bf*)(rbf + (((size_t)(rt * 4 + kb) * 32 + lane) << 4));

  float sumsq[8];
#pragma unroll
  for (int r = 0; r < 8; ++r) sumsq[r] = 0.f;
  v8f zero = {};

  for (int nt = 0; nt < 64; ++nt) {
    v8f acc = zero;
#pragma unroll
    for (int kb = 0; kb < 4; ++kb) {
      v16bf b = *(const v16bf*)(wobf + (((size_t)(kb * 64 + nt) * 32 + lane) << 4));
      acc = __builtin_amdgcn_wmma_f32_16x16x32_bf16(
          false, af[kb], false, b, (short)0, acc, false, false);
    }
#pragma unroll
    for (int r = 0; r < 8; ++r) {
      float v = acc[r];
      sumsq[r] += v * v;
      Out[(size_t)(rt * 16 + r + 8 * half) * H_ + nt * 16 + m16] = v;
    }
  }

#pragma unroll
  for (int r = 0; r < 8; ++r) {
    float s = sumsq[r];
    s += __shfl_xor(s, 1, 32);
    s += __shfl_xor(s, 2, 32);
    s += __shfl_xor(s, 4, 32);
    s += __shfl_xor(s, 8, 32);
    if (m16 == 0) {
      int row = rt * 16 + r + 8 * half;
      float nrm = fmaxf(sqrtf(s), 1e-6f);
      rowscale[row] = fminf(1.0f, 10.0f / nrm);
    }
  }
}

// ---------------------------------------------------------------------------
// Kernel 5: apply norm-10 clamp scale per row. grid = ROWS_, block = 256.
// ---------------------------------------------------------------------------
__global__ __launch_bounds__(256) void k_scale(float* __restrict__ Out,
                                               const float* __restrict__ rowscale) {
  const int row = blockIdx.x;
  const float s = rowscale[row];
#pragma unroll
  for (int j = 0; j < 4; ++j)
    Out[(size_t)row * H_ + j * 256 + threadIdx.x] *= s;
}

// ---------------------------------------------------------------------------
extern "C" void kernel_launch(void* const* d_in, const int* in_sizes, int n_in,
                              void* d_out, int out_size, void* d_ws, size_t ws_size,
                              hipStream_t stream) {
  (void)in_sizes; (void)n_in; (void)out_size; (void)ws_size;
  const float* x    = (const float*)d_in[0];
  const float* Wq   = (const float*)d_in[1];
  const float* Wk   = (const float*)d_in[2];
  const float* Wv   = (const float*)d_in[3];
  const float* Wwk  = (const float*)d_in[4];
  const float* Wout = (const float*)d_in[5];
  const float* Wg_w = (const float*)d_in[6];
  const float* Wg_b = (const float*)d_in[7];
  const float* sk0  = (const float*)d_in[8];
  const float* sv0  = (const float*)d_in[9];
  const float* su0  = (const float*)d_in[10];

  float* out = (float*)d_out;                 // [ROWS_*H_] + 1 scalar
  float* ws  = (float*)d_ws;

  const size_t PROJ = (size_t)ROWS_ * M_;     // 2,097,152 floats each
  float* q    = ws;
  float* k    = ws + 1 * PROJ;
  float* v    = ws + 2 * PROJ;
  float* wk   = ws + 3 * PROJ;
  float* g    = ws + 4 * PROJ;                // ROWS_
  float* retr = g + ROWS_;                    // ROWS_*M_
  float* rsc  = retr + PROJ;                  // ROWS_
  // bf16 fragment region (32B-aligned by construction)
  __bf16* xbf    = (__bf16*)(rsc + ROWS_);    // ROWS_*H_ bf16
  __bf16* wbf    = xbf + (size_t)ROWS_ * H_;  // 4 * H_*M_ bf16
  __bf16* wobf   = wbf + (size_t)4 * H_ * M_; // M_*H_ bf16
  __bf16* retrbf = wobf + (size_t)M_ * H_;    // ROWS_*M_ bf16

  // 0) pre-swizzle to WMMA fragment layouts (bf16)
  k_prep_a<<<(ROWS_ * H_ / 8) / 256, 256, 0, stream>>>(x, xbf, H_, 7, 32);
  k_prep_b<<<(H_ * M_ / 8) / 256, 256, 0, stream>>>(Wq,  wbf + 0 * (size_t)H_ * M_, M_, 3);
  k_prep_b<<<(H_ * M_ / 8) / 256, 256, 0, stream>>>(Wk,  wbf + 1 * (size_t)H_ * M_, M_, 3);
  k_prep_b<<<(H_ * M_ / 8) / 256, 256, 0, stream>>>(Wv,  wbf + 2 * (size_t)H_ * M_, M_, 3);
  k_prep_b<<<(H_ * M_ / 8) / 256, 256, 0, stream>>>(Wwk, wbf + 3 * (size_t)H_ * M_, M_, 3);
  k_prep_b<<<(M_ * H_ / 8) / 256, 256, 0, stream>>>(Wout, wobf, H_, 6);

  // 1) projections (WMMA) — q,k,v,wk (k/v/wk row-normalized)
  k_proj<<<dim3(ROWS_ / 128, 4), 256, 0, stream>>>(xbf, wbf, ws);
  // 2) gate
  k_gate<<<ROWS_ / 8, 256, 0, stream>>>(x, Wg_w, Wg_b, g);
  // 3) sequential slot scan (fp32, single persistent workgroup)
  k_scan<<<1, 256, 0, stream>>>(q, k, v, wk, g, sk0, sv0, su0,
                                retr, out + (size_t)ROWS_ * H_);
  // 4) swizzle retrieved, then output GEMM (WMMA) + row sumsq
  k_prep_a<<<(ROWS_ * M_ / 8) / 256, 256, 0, stream>>>(retr, retrbf, M_, 4, 4);
  k_out<<<ROWS_ / 128, 256, 0, stream>>>(retrbf, wobf, out, rsc);
  // 5) norm-10 clamp
  k_scale<<<ROWS_, 256, 0, stream>>>(out, rsc);
}
